// Model_16492674417055
// MI455X (gfx1250) — matrix operand order repaired and checked by reading, not compiled, not run on hardware
//
#include <hip/hip_runtime.h>
#include <stddef.h>

// ---------------------------------------------------------------------------
// Capsule dynamic routing for MI455X (gfx1250, wave32, WMMA + async-to-LDS).
// Sizes fixed by the reference: B=16, I=1024 (n_pre), J=32 (n_digit), C=4, D=128
// ---------------------------------------------------------------------------

#define B_   16
#define I_   1024
#define J_   32
#define C_   4
#define D_   128
#define M_   (B_ * I_)          // 16384 rows of the big GEMM

typedef __attribute__((ext_vector_type(16))) __bf16 v16bf;
typedef __attribute__((ext_vector_type(8)))  float  v8f;
typedef __attribute__((address_space(3))) unsigned short lds_us_t;

// ----------------------------- bf16 helpers --------------------------------
__device__ __forceinline__ unsigned short f32_to_bf16(float f) {
  unsigned int u = __float_as_uint(f);
  unsigned int r = ((u >> 16) & 1u) + 0x7FFFu;   // round-to-nearest-even
  return (unsigned short)((u + r) >> 16);
}
__device__ __forceinline__ float bf_lo(unsigned int v) {
  return __uint_as_float(v << 16);
}
__device__ __forceinline__ float bf_hi(unsigned int v) {
  return __uint_as_float(v & 0xFFFF0000u);
}

// 32-byte async global->LDS copy (two B128 ops; INST_OFFSET applies to both
// the LDS and global address per CDNA5 ISA §15.18.3). Tracked by ASYNCcnt.
__device__ __forceinline__ void async_copy32(unsigned lds_off,
                                             const unsigned short* gptr) {
  unsigned long long ga = (unsigned long long)(size_t)gptr;
  asm volatile("global_load_async_to_lds_b128 %0, %1, off\n\t"
               "global_load_async_to_lds_b128 %0, %1, off offset:16"
               :: "v"(lds_off), "v"(ga) : "memory");
}

// ------------------------- fp32 -> bf16 convert ----------------------------
__global__ __launch_bounds__(256) void caps_cvt_bf16(
    const float* __restrict__ src, unsigned short* __restrict__ dst, int n) {
  int idx = blockIdx.x * 256 + threadIdx.x;
  if (idx < n) dst[idx] = f32_to_bf16(src[idx]);
}

// ------------------------------ WMMA GEMM ----------------------------------
// u_hat[b,i,c,j,p] = sum_q w[j,c,p,q] * x[b,i,c,q]      (bf16 in, f32 acc, bf16 out)
// A = xbf  [m, c, q]   (m = b*1024+i), row q-contiguous
// B = wbf  [j, c, p, q] -> B-matrix[k=q, n=p]: lane reads w row p over contiguous q
// grid = (M/128, J*C);  block = 256 = 8 waves; wave tile = 32(M) x 64(N)
// W K-slabs are double-buffered in LDS via GLOBAL_LOAD_ASYNC_TO_LDS_B128.
__global__ __launch_bounds__(256) void caps_gemm(
    const unsigned short* __restrict__ xbf,
    const unsigned short* __restrict__ wbf,
    unsigned short* __restrict__ u) {
  const int mtile = blockIdx.x;         // 128-row block tile
  const int jc    = blockIdx.y;         // j*4 + c
  const int j     = jc >> 2;
  const int c     = jc & 3;
  const int tid   = threadIdx.x;
  const int wave  = tid >> 5;
  const int lane  = tid & 31;
  const int mg    = wave >> 1;          // 0..3 : which 32-row band
  const int ng    = wave & 1;           // 0..1 : which 64-col band
  const int lmod  = lane & 15;
  const int lhalf = lane >> 4;          // 0/1

  // double-buffered B slab: [buf][ntile][lane][16 bf16] = 2 x 8 KB
  __shared__ __align__(32) unsigned short ldsB[2][8 * 32 * 16];

  // per-thread staging slot (fixed across K steps)
  const int s_nt = tid >> 5;                       // 0..7  : n-tile
  const int s_bl = tid & 31;                       // frag lane slot
  const int s_n  = s_nt * 16 + (s_bl & 15);        // output column p
  const int s_qh = (s_bl >> 4) * 16;               // q half within 32-K slab
  const unsigned short* s_gsrc =
      wbf + ((size_t)jc * 128 + s_n) * 128 + s_qh; // + K0 per stage
  const unsigned s_slot = (unsigned)((s_nt * 32 + s_bl) * 16) * 2u; // bytes
  const unsigned ldsbase0 =
      (unsigned)(unsigned long long)(lds_us_t*)&ldsB[0][0];
  const unsigned ldsbase1 =
      (unsigned)(unsigned long long)(lds_us_t*)&ldsB[1][0];

  v8f acc[2][4] = {};                   // 2 m-bands x 4 n-tiles, f32 16x16 accum

  // prologue: stage K-slab 0 into buffer 0
  async_copy32(ldsbase0 + s_slot, s_gsrc);

#pragma unroll
  for (int kt = 0; kt < 4; ++kt) {
    const int K0  = kt * 32;
    const int cur = kt & 1;
    // issue next stage into the other buffer (its last readers finished
    // before the barrier that closed iteration kt-1), then wait for stage kt.
    if (kt + 1 < 4) {
      async_copy32((((kt + 1) & 1) ? ldsbase1 : ldsbase0) + s_slot,
                   s_gsrc + (kt + 1) * 32);
      asm volatile("s_wait_asynccnt 0x2" ::: "memory");  // stage kt resident
    } else {
      asm volatile("s_wait_asynccnt 0x0" ::: "memory");
    }
    __syncthreads();   // stage kt visible to all waves

    // ---- A fragments (16x32 bf16): lanes 0-15 hold K 0-15, 16-31 hold 16-31
    v16bf afrag[2];
    for (int mb = 0; mb < 2; ++mb) {
      const int m  = mtile * 128 + mg * 32 + mb * 16 + lmod;
      const int q0 = K0 + lhalf * 16;
      const unsigned short* ap = xbf + ((size_t)m * 4 + c) * 128;
      struct __align__(32) AB { uint4 lo, hi; } tmp;
      tmp.lo = *reinterpret_cast<const uint4*>(ap + q0);
      tmp.hi = *reinterpret_cast<const uint4*>(ap + q0 + 8);
      afrag[mb] = *reinterpret_cast<const v16bf*>(&tmp);
    }

    // ---- 8 WMMAs per K step per wave ----
    for (int nt = 0; nt < 4; ++nt) {
      v16bf bfrag = *reinterpret_cast<const v16bf*>(
          &ldsB[cur][((ng * 4 + nt) * 32 + lane) * 16]);
      for (int mb = 0; mb < 2; ++mb) {
        acc[mb][nt] = __builtin_amdgcn_wmma_f32_16x16x32_bf16(
            false, afrag[mb], false, bfrag, (short)0, acc[mb][nt], false, false);
      }
    }
    __syncthreads();   // all waves done reading buf[cur] -> reusable
  }

  // ---- epilogue: D layout: VGPR v -> row v (lanes 0-15) / v+8 (lanes 16-31), col = lane%16
  for (int mb = 0; mb < 2; ++mb) {
    const int mrow0 = mtile * 128 + mg * 32 + mb * 16 + lhalf * 8;
    for (int nt = 0; nt < 4; ++nt) {
      const int p = ng * 64 + nt * 16 + lmod;
      for (int vv = 0; vv < 8; ++vv) {
        const size_t m = (size_t)(mrow0 + vv);
        u[((m * 4 + c) * 32 + j) * 128 + p] = f32_to_bf16(acc[mb][nt][vv]);
      }
    }
  }
}

// ---------------------- weighted sum over i + squash -----------------------
// s[b,j,c,p] = sum_i cw[b,i,c,j] * u[b,i,c,j,p];  v = squash(s)
// grid = B*J*C = 2048 blocks (bid = b*128 + j*4 + c); block = 256 (8 waves over i)
__global__ __launch_bounds__(256) void caps_wsum(
    const unsigned short* __restrict__ u,
    const float* __restrict__ cw,
    float* __restrict__ vout,
    int uniform) {
  const int bid  = blockIdx.x;
  const int c    = bid & 3;
  const int j    = (bid >> 2) & 31;
  const int b    = bid >> 7;
  const int tid  = threadIdx.x;
  const int wv   = tid >> 5;
  const int lane = tid & 31;

  float a0 = 0.f, a1 = 0.f, a2 = 0.f, a3 = 0.f;
  const int i0 = wv * 128;
  for (int i = i0; i < i0 + 128; ++i) {
    const size_t m  = (size_t)b * 1024 + i;
    const size_t rb = ((m * 4 + c) * 32 + j) * 128;
    __builtin_prefetch(u + rb + (size_t)8 * 4 * 32 * 128, 0, 0);  // 8 rows ahead
    const uint2 pk = *reinterpret_cast<const uint2*>(u + rb + lane * 4);
    const float wgt = uniform ? (1.0f / 32.0f) : cw[(m * 4 + c) * 32 + j];
    a0 += wgt * bf_lo(pk.x);
    a1 += wgt * bf_hi(pk.x);
    a2 += wgt * bf_lo(pk.y);
    a3 += wgt * bf_hi(pk.y);
  }

  __shared__ float red[8][128];
  red[wv][lane * 4 + 0] = a0;
  red[wv][lane * 4 + 1] = a1;
  red[wv][lane * 4 + 2] = a2;
  red[wv][lane * 4 + 3] = a3;
  __syncthreads();

  if (wv == 0) {
    float s0 = 0.f, s1 = 0.f, s2 = 0.f, s3 = 0.f;
    for (int ww = 0; ww < 8; ++ww) {
      s0 += red[ww][lane * 4 + 0];
      s1 += red[ww][lane * 4 + 1];
      s2 += red[ww][lane * 4 + 2];
      s3 += red[ww][lane * 4 + 3];
    }
    float sq = s0 * s0 + s1 * s1 + s2 * s2 + s3 * s3;
    for (int mask = 16; mask > 0; mask >>= 1) sq += __shfl_xor(sq, mask, 32);
    const float scale = (sq / (1.0f + sq)) * rsqrtf(sq + 1e-7f);
    float* dst = vout + (((size_t)b * 32 + j) * 4 + c) * 128 + lane * 4;
    dst[0] = s0 * scale;
    dst[1] = s1 * scale;
    dst[2] = s2 * scale;
    dst[3] = s3 * scale;
  }
}

// ---------------------------- agreement pass -------------------------------
// blog[b,i,c,j] (+)= sum_p u[b,i,c,j,p] * v[b,j,c,p]
// grid = B*C*J = 2048 (bid = b*128 + c*32 + j); block = 256, 4 i's per thread
__global__ __launch_bounds__(256) void caps_agree(
    const unsigned short* __restrict__ u,
    const float* __restrict__ v,
    float* __restrict__ blog,
    int init) {
  const int bid = blockIdx.x;
  const int j   = bid & 31;
  const int c   = (bid >> 5) & 3;
  const int b   = bid >> 7;
  const int tid = threadIdx.x;

  __shared__ float vj[128];
  if (tid < 128) vj[tid] = v[(((size_t)b * 32 + j) * 4 + c) * 128 + tid];
  __syncthreads();

  for (int rep = 0; rep < 4; ++rep) {
    const int i = rep * 256 + tid;
    const size_t m = (size_t)b * 1024 + i;
    const unsigned short* row = u + ((m * 4 + c) * 32 + j) * 128;
    float dot = 0.f;
    for (int p = 0; p < 128; p += 8) {
      const uint4 pk = *reinterpret_cast<const uint4*>(row + p);
      dot += bf_lo(pk.x) * vj[p + 0] + bf_hi(pk.x) * vj[p + 1];
      dot += bf_lo(pk.y) * vj[p + 2] + bf_hi(pk.y) * vj[p + 3];
      dot += bf_lo(pk.z) * vj[p + 4] + bf_hi(pk.z) * vj[p + 5];
      dot += bf_lo(pk.w) * vj[p + 6] + bf_hi(pk.w) * vj[p + 7];
    }
    const size_t bi = (m * 4 + c) * 32 + j;
    blog[bi] = init ? dot : (blog[bi] + dot);
  }
}

// ------------------------- softmax over j (axis=2) -------------------------
// cw[b,i,c,:] = softmax(blog[b,i,c,:]); grid = 65536/256 = 256 blocks
__global__ __launch_bounds__(256) void caps_softmax(
    const float* __restrict__ blog, float* __restrict__ cw) {
  const int idx = blockIdx.x * 256 + threadIdx.x;   // (b,i,c) flat, 0..65535
  const float* src = blog + (size_t)idx * 32;
  float e[32];
  float mx = src[0];
  for (int jj = 1; jj < 32; ++jj) mx = fmaxf(mx, src[jj]);
  float s = 0.f;
  for (int jj = 0; jj < 32; ++jj) { e[jj] = __expf(src[jj] - mx); s += e[jj]; }
  const float inv = 1.0f / s;
  float* dst = cw + (size_t)idx * 32;
  for (int jj = 0; jj < 32; ++jj) dst[jj] = e[jj] * inv;
}

// ------------------------------- launcher ----------------------------------
extern "C" void kernel_launch(void* const* d_in, const int* in_sizes, int n_in,
                              void* d_out, int out_size, void* d_ws, size_t ws_size,
                              hipStream_t stream) {
  (void)in_sizes; (void)n_in; (void)out_size; (void)ws_size;
  const float* x = (const float*)d_in[0];   // [16,1024,4,128]  = 8388608 f32
  const float* w = (const float*)d_in[1];   // [32,4,128,128]   = 2097152 f32
  float* out = (float*)d_out;               // [16,32,4,128]    = 262144 f32

  // workspace carve-up
  constexpr size_t U_BYTES   = (size_t)M_ * C_ * J_ * D_ * 2;   // 512 MiB bf16 u_hat
  constexpr size_t XB_BYTES  = (size_t)M_ * C_ * D_ * 2;        // 16 MiB
  constexpr size_t WB_BYTES  = (size_t)J_ * C_ * D_ * D_ * 2;   // 4 MiB
  constexpr size_t BL_BYTES  = (size_t)M_ * C_ * J_ * 4;        // 8 MiB logits
  constexpr size_t CW_BYTES  = BL_BYTES;                        // 8 MiB softmax
  char* ws = (char*)d_ws;
  unsigned short* u_hat = (unsigned short*)ws;
  unsigned short* xbf   = (unsigned short*)(ws + U_BYTES);
  unsigned short* wbf   = (unsigned short*)(ws + U_BYTES + XB_BYTES);
  float* blog = (float*)(ws + U_BYTES + XB_BYTES + WB_BYTES);
  float* cw   = (float*)(ws + U_BYTES + XB_BYTES + WB_BYTES + BL_BYTES);
  float* vbuf = (float*)(ws + U_BYTES + XB_BYTES + WB_BYTES + BL_BYTES + CW_BYTES);

  const int nx = M_ * C_ * D_;           // 8388608
  const int nw = J_ * C_ * D_ * D_;      // 2097152
  caps_cvt_bf16<<<(nx + 255) / 256, 256, 0, stream>>>(x, xbf, nx);
  caps_cvt_bf16<<<(nw + 255) / 256, 256, 0, stream>>>(w, wbf, nw);

  // u_hat = x @ w^T per (j,c), bf16 WMMA with async double-buffered B slabs
  caps_gemm<<<dim3(M_ / 128, J_ * C_), 256, 0, stream>>>(xbf, wbf, u_hat);

  const int nblk = B_ * J_ * C_;         // 2048
  // iteration 0: uniform coupling c = 1/32
  caps_wsum<<<nblk, 256, 0, stream>>>(u_hat, cw, vbuf, /*uniform=*/1);
  // iteration 1
  caps_agree<<<nblk, 256, 0, stream>>>(u_hat, vbuf, blog, /*init=*/1);
  caps_softmax<<<(M_ * C_) / 256, 256, 0, stream>>>(blog, cw);
  caps_wsum<<<nblk, 256, 0, stream>>>(u_hat, cw, vbuf, /*uniform=*/0);
  // iteration 2 (final) -> writes v into d_out
  caps_agree<<<nblk, 256, 0, stream>>>(u_hat, vbuf, blog, /*init=*/0);
  caps_softmax<<<(M_ * C_) / 256, 256, 0, stream>>>(blog, cw);
  caps_wsum<<<nblk, 256, 0, stream>>>(u_hat, cw, out, /*uniform=*/0);
}
